// Point_Transformer_Last_1133871366468
// MI455X (gfx1250) — compile-verified, matrix operand order checked by reading, not run on hardware
//
#include <hip/hip_runtime.h>
#include <stdint.h>

#define B_ 16
#define C_ 256
#define N_ 2048
#define QD 64
#define NUM_SA 4
#define EPS_BN 1e-3f

typedef __attribute__((ext_vector_type(16))) __bf16 v16bf;
typedef __attribute__((ext_vector_type(8)))  float  v8f;

union FragU { v16bf v; uint32_t u[8]; };

__device__ __forceinline__ v8f zero8() {
  v8f z = {0.f, 0.f, 0.f, 0.f, 0.f, 0.f, 0.f, 0.f};
  return z;
}

__device__ __forceinline__ uint16_t f2bfu(float f) {
  union { float f; uint32_t u; } a; a.f = f;
  uint32_t r = a.u + 0x7FFFu + ((a.u >> 16) & 1u);   // round-to-nearest-even
  return (uint16_t)(r >> 16);
}
__device__ __forceinline__ uint32_t pack2bf(float lo, float hi) {
  return (uint32_t)f2bfu(lo) | ((uint32_t)f2bfu(hi) << 16);
}

// ---- CDNA5 async memory->LDS copy (16B per lane), tracked by ASYNCcnt ----
__device__ __forceinline__ void async_copy_b128(const __bf16* lds_dst,
                                                const __bf16* gsrc) {
  uint32_t lds_off = (uint32_t)(uintptr_t)lds_dst;  // low 32b of generic = LDS offset
  asm volatile("global_load_async_to_lds_b128 %0, %1, off"
               :: "v"(lds_off), "v"(gsrc) : "memory");
}
__device__ __forceinline__ void wait_async_le2() {
  asm volatile("s_wait_asynccnt 0x2" ::: "memory");
}
__device__ __forceinline__ void wait_async_le0() {
  asm volatile("s_wait_asynccnt 0x0" ::: "memory");
}

// Load a 16x32 bf16 fragment (A or B^T) from a row-major-over-K buffer.
// lane_row must already point at this lane's row (row index = lane & 15).
// ISA 16-bit A/B layout: lanes 0-15 half=0, lanes 16-31 half=1;
// dword v holds K pair at kk = (v/4)*16 + half*8 + (v%4)*2.
__device__ __forceinline__ v16bf load_frag(const __bf16* lane_row, int k0) {
  const int half = (threadIdx.x >> 4) & 1;
  FragU f;
  const uint32_t* p = (const uint32_t*)lane_row;
#pragma unroll
  for (int v = 0; v < 8; ++v) {
    const int kk = ((v >> 2) << 4) + (half << 3) + ((v & 3) << 1);
    f.u[v] = p[(k0 + kk) >> 1];
  }
  return f.v;
}

__device__ __forceinline__ v8f wmma_bf16(v16bf a, v16bf b, v8f c) {
  return __builtin_amdgcn_wmma_f32_16x16x32_bf16(false, a, false, b, (short)0, c,
                                                 false, false);
}

// ---------------------------------------------------------------------------
// 4-wave (128 thread) workgroup GEMM core: 64 co rows x 64 n cols.
// Wave w computes rows [co_base + 16w, +16). The B operand (inT, 64 rows x 32 K
// per step) is shared by all 4 waves: staged into LDS with async-to-LDS copies,
// double-buffered, drained with s_wait_asynccnt.
// A: W [Cout][256] bf16 row-major (global, small, L2-resident).
// Bt: inT [N][256] bf16 row-major-over-K.
// ---------------------------------------------------------------------------
__device__ __forceinline__ void gemm_block64(const __bf16* W, int co_base,
                                             const __bf16* inT_blk, v8f acc[4]) {
  __shared__ __bf16 Bs[2][64 * 32];   // 8 KB double-buffered slab
  const int tid = threadIdx.x;
  const int wave = tid >> 5;
  const int lane = tid & 31, m = lane & 15;
  const __bf16* Arow = W + (size_t)(co_base + wave * 16 + m) * C_;

  auto stage = [&](int k0, int buf) {
#pragma unroll
    for (int c2 = 0; c2 < 2; ++c2) {
      const int chunk = tid + c2 * 128;   // 256 chunks of 16B = 64 rows x 64B
      const int row = chunk >> 2;
      const int off = (chunk & 3) * 8;    // element offset within row (8 bf16 = 16B)
      async_copy_b128(&Bs[buf][row * 32 + off],
                      inT_blk + (size_t)row * C_ + k0 + off);
    }
  };

  stage(0, 0);
  for (int k0 = 0; k0 < C_; k0 += 32) {
    const int buf = (k0 >> 5) & 1;
    if (k0 + 32 < C_) {
      stage(k0 + 32, buf ^ 1);  // prefetch next slab into other buffer
      wait_async_le2();         // current slab's 2 copies done; next 2 in flight
    } else {
      wait_async_le0();
    }
    __syncthreads();
    v16bf a = load_frag(Arow, k0);
#pragma unroll
    for (int t = 0; t < 4; ++t) {
      v16bf bb = load_frag(&Bs[buf][(t * 16 + m) * 32], 0);
      acc[t] = wmma_bf16(a, bb, acc[t]);
    }
    __syncthreads();            // slab consumed before it is overwritten
  }
}

// ---------------- setup kernels ----------------

__global__ void k_f32_to_bf16(const float* __restrict__ src,
                              __bf16* __restrict__ dst, int n) {
  int i = blockIdx.x * blockDim.x + threadIdx.x;
  const int stride = gridDim.x * blockDim.x;
  uint16_t* d = (uint16_t*)dst;
  for (; i < n; i += stride) d[i] = f2bfu(src[i]);
}

// x [B][C][N] f32 -> xT [B][N][C] bf16
__global__ void k_x_to_xT(const float* __restrict__ x, __bf16* __restrict__ xT) {
  const int total = B_ * C_ * N_;
  int i = blockIdx.x * blockDim.x + threadIdx.x;
  const int stride = gridDim.x * blockDim.x;
  uint16_t* d = (uint16_t*)xT;
  for (; i < total; i += stride) {
    int b = i / (C_ * N_);
    int r = i - b * (C_ * N_);
    int c = r / N_;
    int n = r - c * N_;
    d[(size_t)b * N_ * C_ + (size_t)n * C_ + c] = f2bfu(x[i]);
  }
}

// ---------------- GEMM kernels (128-thread blocks) ----------------

// out[co][n] = relu(bn(W @ in)); writes outT [B][N][C] bf16 and optional f32 [B][C][N]
__global__ void k_conv_bn_relu(const __bf16* __restrict__ W,
                               const __bf16* __restrict__ inT,
                               const float* __restrict__ g, const float* __restrict__ bb,
                               const float* __restrict__ mm, const float* __restrict__ vv,
                               __bf16* __restrict__ outT, float* __restrict__ outF) {
  const int n0 = blockIdx.x * 64, co_base = blockIdx.y * 64, b = blockIdx.z;
  v8f acc[4];
#pragma unroll
  for (int t = 0; t < 4; ++t) acc[t] = zero8();
  gemm_block64(W, co_base, inT + ((size_t)b * N_ + n0) * C_, acc);

  const int lane = threadIdx.x & 31, wave = threadIdx.x >> 5;
  const int m = lane & 15, half = lane >> 4;
  const int co0 = co_base + wave * 16;
#pragma unroll
  for (int t = 0; t < 4; ++t) {
    const int n = n0 + t * 16 + m;
#pragma unroll
    for (int v = 0; v < 8; v += 2) {
      float r01[2];
#pragma unroll
      for (int s = 0; s < 2; ++s) {
        const int co = co0 + half * 8 + v + s;
        const float sc = g[co] * rsqrtf(vv[co] + EPS_BN);
        float val = (acc[t][v + s] - mm[co]) * sc + bb[co];
        val = fmaxf(val, 0.f);
        r01[s] = val;
        if (outF) outF[((size_t)b * C_ + co) * N_ + n] = val;
      }
      *(uint32_t*)((uint16_t*)outT + ((size_t)b * N_ + n) * C_ + co0 + half * 8 + v) =
          pack2bf(r01[0], r01[1]);
    }
  }
}

// qT[b][n][o] = (Wq @ h)[o][n]   (Cout = 64 -> one co-group)
__global__ void k_q(const __bf16* __restrict__ Wq, const __bf16* __restrict__ hT,
                    __bf16* __restrict__ qT) {
  const int n0 = blockIdx.x * 64, b = blockIdx.z;
  v8f acc[4];
#pragma unroll
  for (int t = 0; t < 4; ++t) acc[t] = zero8();
  gemm_block64(Wq, 0, hT + ((size_t)b * N_ + n0) * C_, acc);

  const int lane = threadIdx.x & 31, wave = threadIdx.x >> 5;
  const int m = lane & 15, half = lane >> 4;
  const int o0 = wave * 16;
#pragma unroll
  for (int t = 0; t < 4; ++t) {
    const int n = n0 + t * 16 + m;
#pragma unroll
    for (int v = 0; v < 8; v += 2) {
      *(uint32_t*)((uint16_t*)qT + ((size_t)b * N_ + n) * QD + o0 + half * 8 + v) =
          pack2bf(acc[t][v], acc[t][v + 1]);
    }
  }
}

// xv[b][c][n] = (Wv @ h)[c][n] + vb[c]   (bf16, [C][N] layout: A-operand for xv@S)
__global__ void k_xv(const __bf16* __restrict__ Wv, const float* __restrict__ vb,
                     const __bf16* __restrict__ hT, __bf16* __restrict__ xv) {
  const int n0 = blockIdx.x * 64, co_base = blockIdx.y * 64, b = blockIdx.z;
  v8f acc[4];
#pragma unroll
  for (int t = 0; t < 4; ++t) acc[t] = zero8();
  gemm_block64(Wv, co_base, hT + ((size_t)b * N_ + n0) * C_, acc);

  const int lane = threadIdx.x & 31, wave = threadIdx.x >> 5;
  const int m = lane & 15, half = lane >> 4;
  const int co0 = co_base + wave * 16;
  uint16_t* dst = (uint16_t*)xv;
#pragma unroll
  for (int t = 0; t < 4; ++t) {
    const int n = n0 + t * 16 + m;
#pragma unroll
    for (int v = 0; v < 8; ++v) {
      const int co = co0 + half * 8 + v;
      dst[((size_t)b * C_ + co) * N_ + n] = f2bfu(acc[t][v] + vb[co]);
    }
  }
}

// t = Wt @ u + tb ; newh = h + relu(bn(t)); updates hF (in place), hT, d_out slice
__global__ void k_tconv(const __bf16* __restrict__ Wt, const float* __restrict__ tb,
                        const float* __restrict__ g, const float* __restrict__ bb,
                        const float* __restrict__ mm, const float* __restrict__ vv,
                        const __bf16* __restrict__ uT, float* __restrict__ hF,
                        __bf16* __restrict__ hT, float* __restrict__ out, int layer) {
  const int n0 = blockIdx.x * 64, co_base = blockIdx.y * 64, b = blockIdx.z;
  v8f acc[4];
#pragma unroll
  for (int t = 0; t < 4; ++t) acc[t] = zero8();
  gemm_block64(Wt, co_base, uT + ((size_t)b * N_ + n0) * C_, acc);

  const int lane = threadIdx.x & 31, wave = threadIdx.x >> 5;
  const int m = lane & 15, half = lane >> 4;
  const int co0 = co_base + wave * 16;
#pragma unroll
  for (int t = 0; t < 4; ++t) {
    const int n = n0 + t * 16 + m;
#pragma unroll
    for (int v = 0; v < 8; v += 2) {
      float nh01[2];
#pragma unroll
      for (int s = 0; s < 2; ++s) {
        const int co = co0 + half * 8 + v + s;
        const float tval = acc[t][v + s] + tb[co];
        const float sc = g[co] * rsqrtf(vv[co] + EPS_BN);
        const float z = fmaxf((tval - mm[co]) * sc + bb[co], 0.f);
        const float nh = hF[((size_t)b * C_ + co) * N_ + n] + z;
        hF[((size_t)b * C_ + co) * N_ + n] = nh;
        out[((size_t)b * (4 * C_) + layer * C_ + co) * N_ + n] = nh;
        nh01[s] = nh;
      }
      *(uint32_t*)((uint16_t*)hT + ((size_t)b * N_ + n) * C_ + co0 + half * 8 + v) =
          pack2bf(nh01[0], nh01[1]);
    }
  }
}

// ---------------- attention kernels (1-wave blocks) ----------------

// Pass 1: per-row max & sum-exp of energy = qT qT^T (online, flash-style)
__global__ void k_rowstats(const __bf16* __restrict__ qT,
                           float* __restrict__ rowmax, float* __restrict__ rowsum) {
  const int n0 = blockIdx.x * 16, b = blockIdx.y;
  __shared__ float e_lds[16][17];
  const int lane = threadIdx.x, m = lane & 15, half = lane >> 4;
  const __bf16* qb = qT + (size_t)b * N_ * QD;
  float rm = -3.0e38f, rs = 0.f;

  for (int m0 = 0; m0 < N_; m0 += 16) {
    if (m0 + 16 < N_)
      __builtin_prefetch(qb + (size_t)(m0 + 16 + lane) * QD, 0, 1);
    v8f e = zero8();
#pragma unroll
    for (int k0 = 0; k0 < QD; k0 += 32) {
      v16bf a = load_frag(qb + (size_t)(n0 + m) * QD, k0);
      v16bf bf = load_frag(qb + (size_t)(m0 + m) * QD, k0);
      e = wmma_bf16(a, bf, e);
    }
#pragma unroll
    for (int v = 0; v < 8; ++v) e_lds[v + 8 * half][m] = e[v];
    __syncthreads();
    if (lane < 16) {
      float cmax = rm;
      for (int c2 = 0; c2 < 16; ++c2) cmax = fmaxf(cmax, e_lds[lane][c2]);
      const float scale = __expf(rm - cmax);
      float add = 0.f;
      for (int c2 = 0; c2 < 16; ++c2) add += __expf(e_lds[lane][c2] - cmax);
      rs = rs * scale + add;
      rm = cmax;
    }
    __syncthreads();
  }
  if (lane < 16) {
    rowmax[(size_t)b * N_ + n0 + lane] = rm;
    rowsum[(size_t)b * N_ + n0 + lane] = rs;
  }
}

// Pass 2: recompute E tiles, S = softmax row-normalized (bf16 in LDS, consumed
// directly as WMMA B operand), y = xv @ S, colsum tracked per column, then
// u = h - y/(1e-9+colsum), stored transposed as uT [B][N][C] bf16.
__global__ void k_attn_apply(const __bf16* __restrict__ qT, const __bf16* __restrict__ xv,
                             const float* __restrict__ hF,
                             const float* __restrict__ rowmax,
                             const float* __restrict__ rowsum,
                             __bf16* __restrict__ uT) {
  const int m0 = blockIdx.x * 16, c0 = blockIdx.y * 64, b = blockIdx.z;
  __shared__ uint32_t S_pack[16][16];  // [m-col][k-pair], bf16 x2 per dword
  const int lane = threadIdx.x, m = lane & 15, half = lane >> 4;
  const __bf16* qb = qT + (size_t)b * N_ * QD;

  v8f acc[4];
#pragma unroll
  for (int t = 0; t < 4; ++t) acc[t] = zero8();
  float csum = 0.f;

  for (int nb = 0; nb < N_; nb += 32) {
    if (nb + 32 < N_) {
      __builtin_prefetch(qb + (size_t)(nb + 32 + lane) * QD, 0, 1);
      __builtin_prefetch(xv + ((size_t)b * C_ + c0 + 16 * (lane >> 4) + m) * N_ + nb + 32,
                         0, 1);
    }
    const float rm_l = rowmax[(size_t)b * N_ + nb + lane];
    const float rs_l = rowsum[(size_t)b * N_ + nb + lane];
#pragma unroll
    for (int sub = 0; sub < 2; ++sub) {
      v8f e = zero8();
#pragma unroll
      for (int k0 = 0; k0 < QD; k0 += 32) {
        v16bf a = load_frag(qb + (size_t)(nb + sub * 16 + m) * QD, k0);
        v16bf bfr = load_frag(qb + (size_t)(m0 + m) * QD, k0);
        e = wmma_bf16(a, bfr, e);
      }
#pragma unroll
      for (int v = 0; v < 8; v += 2) {
        const int rloc = sub * 16 + 8 * half + v;
        const float s0 = __expf(e[v] - __shfl(rm_l, rloc)) / __shfl(rs_l, rloc);
        const float s1 = __expf(e[v + 1] - __shfl(rm_l, rloc + 1)) / __shfl(rs_l, rloc + 1);
        csum += s0 + s1;
        S_pack[m][rloc >> 1] = pack2bf(s0, s1);
      }
    }
    __syncthreads();
    // B fragment of S (shared by all 4 channel tiles)
    FragU bS;
    const uint32_t* sp = &S_pack[m][0];
#pragma unroll
    for (int v = 0; v < 8; ++v) {
      const int kk = ((v >> 2) << 4) + (half << 3) + ((v & 3) << 1);
      bS.u[v] = sp[kk >> 1];
    }
#pragma unroll
    for (int t = 0; t < 4; ++t) {
      v16bf a = load_frag(xv + ((size_t)b * C_ + c0 + t * 16 + m) * N_, nb);
      acc[t] = wmma_bf16(a, bS.v, acc[t]);
    }
    __syncthreads();
  }

  const float ctot = csum + __shfl_xor(csum, 16);
  const float inv = 1.0f / (1e-9f + ctot);
  const int mc = m0 + m;
#pragma unroll
  for (int t = 0; t < 4; ++t) {
#pragma unroll
    for (int v = 0; v < 8; v += 2) {
      const int c = c0 + t * 16 + 8 * half + v;
      const float u0 = hF[((size_t)b * C_ + c) * N_ + mc] - acc[t][v] * inv;
      const float u1 = hF[((size_t)b * C_ + c + 1) * N_ + mc] - acc[t][v + 1] * inv;
      *(uint32_t*)((uint16_t*)uT + ((size_t)b * N_ + mc) * C_ + c) = pack2bf(u0, u1);
    }
  }
}

extern "C" void kernel_launch(void* const* d_in, const int* in_sizes, int n_in,
                              void* d_out, int out_size, void* d_ws, size_t ws_size,
                              hipStream_t stream) {
  (void)in_sizes; (void)n_in; (void)out_size; (void)ws_size;
  const float* x       = (const float*)d_in[0];
  const float* conv1_w = (const float*)d_in[1];
  const float* conv2_w = (const float*)d_in[2];
  const float* bn1_g = (const float*)d_in[3];
  const float* bn1_b = (const float*)d_in[4];
  const float* bn1_m = (const float*)d_in[5];
  const float* bn1_v = (const float*)d_in[6];
  const float* bn2_g = (const float*)d_in[7];
  const float* bn2_b = (const float*)d_in[8];
  const float* bn2_m = (const float*)d_in[9];
  const float* bn2_v = (const float*)d_in[10];
  const float* sa_qk_w = (const float*)d_in[11];
  const float* sa_v_w  = (const float*)d_in[12];
  const float* sa_v_b  = (const float*)d_in[13];
  const float* sa_t_w  = (const float*)d_in[14];
  const float* sa_t_b  = (const float*)d_in[15];
  const float* sa_g    = (const float*)d_in[16];
  const float* sa_b    = (const float*)d_in[17];
  const float* sa_m    = (const float*)d_in[18];
  const float* sa_var  = (const float*)d_in[19];
  float* out = (float*)d_out;

  char* p = (char*)d_ws;
  auto carve = [&](size_t bytes) -> char* {
    char* r = p; p += (bytes + 255) & ~(size_t)255; return r;
  };
  __bf16* w1b  = (__bf16*)carve((size_t)C_ * C_ * 2);
  __bf16* w2b  = (__bf16*)carve((size_t)C_ * C_ * 2);
  __bf16* wqkb = (__bf16*)carve((size_t)NUM_SA * QD * C_ * 2);
  __bf16* wvb  = (__bf16*)carve((size_t)NUM_SA * C_ * C_ * 2);
  __bf16* wtb  = (__bf16*)carve((size_t)NUM_SA * C_ * C_ * 2);
  __bf16* xT   = (__bf16*)carve((size_t)B_ * N_ * C_ * 2);
  __bf16* h1T  = (__bf16*)carve((size_t)B_ * N_ * C_ * 2);
  __bf16* hT   = (__bf16*)carve((size_t)B_ * N_ * C_ * 2);
  float*  hF   = (float*) carve((size_t)B_ * C_ * N_ * 4);
  __bf16* qT   = (__bf16*)carve((size_t)B_ * N_ * QD * 2);
  __bf16* xv   = (__bf16*)carve((size_t)B_ * C_ * N_ * 2);
  __bf16* uT   = (__bf16*)carve((size_t)B_ * N_ * C_ * 2);
  float*  rmx  = (float*) carve((size_t)B_ * N_ * 4);
  float*  rsm  = (float*) carve((size_t)B_ * N_ * 4);

  const dim3 blk128(128);  // 4 waves per WGP-friendly workgroup
  const dim3 blk32(32);
  // weight conversion to bf16
  k_f32_to_bf16<<<dim3(64), dim3(256), 0, stream>>>(conv1_w, w1b, C_ * C_);
  k_f32_to_bf16<<<dim3(64), dim3(256), 0, stream>>>(conv2_w, w2b, C_ * C_);
  k_f32_to_bf16<<<dim3(64), dim3(256), 0, stream>>>(sa_qk_w, wqkb, NUM_SA * QD * C_);
  k_f32_to_bf16<<<dim3(256), dim3(256), 0, stream>>>(sa_v_w, wvb, NUM_SA * C_ * C_);
  k_f32_to_bf16<<<dim3(256), dim3(256), 0, stream>>>(sa_t_w, wtb, NUM_SA * C_ * C_);
  // x -> transposed bf16
  k_x_to_xT<<<dim3(4096), dim3(256), 0, stream>>>(x, xT);

  // conv1 -> h1T, conv2 -> hT + hF
  k_conv_bn_relu<<<dim3(N_ / 64, C_ / 64, B_), blk128, 0, stream>>>(
      w1b, xT, bn1_g, bn1_b, bn1_m, bn1_v, h1T, nullptr);
  k_conv_bn_relu<<<dim3(N_ / 64, C_ / 64, B_), blk128, 0, stream>>>(
      w2b, h1T, bn2_g, bn2_b, bn2_m, bn2_v, hT, hF);

  for (int i = 0; i < NUM_SA; ++i) {
    k_q<<<dim3(N_ / 64, 1, B_), blk128, 0, stream>>>(
        wqkb + (size_t)i * QD * C_, hT, qT);
    k_xv<<<dim3(N_ / 64, C_ / 64, B_), blk128, 0, stream>>>(
        wvb + (size_t)i * C_ * C_, sa_v_b + (size_t)i * C_, hT, xv);
    k_rowstats<<<dim3(N_ / 16, B_), blk32, 0, stream>>>(qT, rmx, rsm);
    k_attn_apply<<<dim3(N_ / 16, C_ / 64, B_), blk32, 0, stream>>>(
        qT, xv, hF, rmx, rsm, uT);
    k_tconv<<<dim3(N_ / 64, C_ / 64, B_), blk128, 0, stream>>>(
        wtb + (size_t)i * C_ * C_, sa_t_b + (size_t)i * C_,
        sa_g + (size_t)i * C_, sa_b + (size_t)i * C_,
        sa_m + (size_t)i * C_, sa_var + (size_t)i * C_,
        uT, hF, hT, out, i);
  }
}
